// SoftShiftAligner_30477087932747
// MI455X (gfx1250) — matrix operand order repaired and checked by reading, not compiled, not run on hardware
//
#include <hip/hip_runtime.h>
#include <hip/hip_bf16.h>

typedef __attribute__((ext_vector_type(16))) _Float16 v16h;
typedef __attribute__((ext_vector_type(8)))  float    v8f;

#define BATCH 4
#define CH    32
#define HH    128
#define WW    128
#define WIN   9
#define PADR  4
#define WP    144               // padded width for channel-major key: 4 + 128 + 12
#define INV_T (1.0f/0.07f)

#define NPIX       (BATCH*HH*WW)        // 65536
#define QN_T_ELEMS (NPIX*CH)            // 2097152
#define NPAD       (BATCH*CH*HH*16)     // pad columns to zero-fill

union F16x16 { v16h h; uint4 u[2]; };

static __device__ inline v16h zero16() {
  v16h v;
#pragma unroll
  for (int i = 0; i < 16; ++i) v[i] = (_Float16)0.0f;
  return v;
}
static __device__ inline v8f zero8() {
  v8f v;
#pragma unroll
  for (int i = 0; i < 8; ++i) v[i] = 0.0f;
  return v;
}
// A-matrix fragment (16x32 f16): lane holds row M, K = v + 16*half (contiguous run)
static __device__ inline v16h loadA(const _Float16* row, int half) {
  F16x16 f;
  f.u[0] = *reinterpret_cast<const uint4*>(row + half*16);
  f.u[1] = *reinterpret_cast<const uint4*>(row + half*16 + 8);
  return f.h;
}
// B-matrix fragment (32x16 f16): lane holds col N, K = v + 16*half (contiguous run)
static __device__ inline v16h loadB(const _Float16* row, int half) {
  F16x16 f;
  f.u[0] = *reinterpret_cast<const uint4*>(row + half*16);
  f.u[1] = *reinterpret_cast<const uint4*>(row + half*16 + 8);
  return f.h;
}

// --------- pass 1: per-pixel L2 normalize, emit f16 in WMMA-friendly layouts ----------
__global__ __launch_bounds__(256)
void SSA_nrm_kernel(const float* __restrict__ q, const float* __restrict__ k,
                    _Float16* __restrict__ qn_t, _Float16* __restrict__ kn_t,
                    _Float16* __restrict__ kn_c) {
  int p = blockIdx.x * 256 + threadIdx.x;
  if (p >= NPIX) return;
  int b = p >> 14, hw = p & 16383;
  int h = hw >> 7, w = hw & 127;
  float qv[CH], kv[CH];
  float qs = 0.f, ks = 0.f;
  int base = ((b * CH) * HH + h) * WW + w;
#pragma unroll
  for (int c = 0; c < CH; ++c) {
    float x = q[base + c * HH * WW]; qv[c] = x; qs += x * x;
    float y = k[base + c * HH * WW]; kv[c] = y; ks += y * y;
  }
  float iq = 1.f / fmaxf(sqrtf(qs), 1e-6f);
  float ik = 1.f / fmaxf(sqrtf(ks), 1e-6f);
  _Float16* qo = qn_t + p * CH;
  _Float16* ko = kn_t + p * CH;
#pragma unroll
  for (int c = 0; c < CH; ++c) {
    qo[c] = (_Float16)(qv[c] * iq);
    ko[c] = (_Float16)(kv[c] * ik);
  }
  int cbase = ((b * CH) * HH + h) * WP + PADR + w;   // channel-major, +4 col halo
#pragma unroll
  for (int c = 0; c < CH; ++c)
    kn_c[cbase + c * HH * WP] = (_Float16)(kv[c] * ik);
}

// --------- zero-fill halo columns of channel-major key ----------
__global__ __launch_bounds__(256)
void SSA_pad_kernel(_Float16* __restrict__ kn_c) {
  int idx = blockIdx.x * 256 + threadIdx.x;
  if (idx >= NPAD) return;
  int p2 = idx & 15;
  int h  = (idx >> 4) & 127;
  int c  = (idx >> 11) & 31;
  int b  = idx >> 16;
  int col = (p2 < 4) ? p2 : (p2 + 128);             // cols 0..3 and 132..143
  kn_c[((b * CH + c) * HH + h) * WP + col] = (_Float16)0.0f;
}

// --------- pass 2: one wave per 16-pixel row segment; WMMA sim + softmax + WMMA blend ----
__global__ __launch_bounds__(32)
void SSA_align_kernel(const _Float16* __restrict__ qn_t,
                      const _Float16* __restrict__ kn_t,
                      const _Float16* __restrict__ kn_c,
                      float* __restrict__ out) {
  __shared__ float slds[2][16][16];   // S-tile dump
  __shared__ float simw[16][WIN * WIN + 3];

  int bid = blockIdx.x;
  int w0 = (bid & 7) << 4;
  int h  = (bid >> 3) & 127;
  int b  = bid >> 10;
  int lane = threadIdx.x;
  int half = lane >> 4, m = lane & 15;

  // Q tile: A[M=pixel i][K=channel]
  v16h aq = loadA(qn_t + ((size_t)((b * HH + h) * WW) + w0 + m) * CH, half);

  // ---- sim phase: logits for all 81 window positions ----
  for (int dy = 0; dy < WIN; ++dy) {
    int r = h + dy - PADR;
    bool rok = (r >= 0) && (r < HH);
    if (rok) {
      v8f s0 = zero8(), s1 = zero8();
      {
        int col = w0 - PADR + m;                      // key tile j0 = 0
        v16h bk = zero16();
        if (col >= 0 && col < WW)
          bk = loadB(kn_t + ((size_t)((b * HH + r) * WW) + col) * CH, half);
        s0 = __builtin_amdgcn_wmma_f32_16x16x32_f16(false, aq, false, bk,
                                                    (short)0, s0, false, false);
      }
      {
        int col = w0 - PADR + 8 + m;                  // key tile j0 = 8
        v16h bk = zero16();
        if (col >= 0 && col < WW)
          bk = loadB(kn_t + ((size_t)((b * HH + r) * WW) + col) * CH, half);
        s1 = __builtin_amdgcn_wmma_f32_16x16x32_f16(false, aq, false, bk,
                                                    (short)0, s1, false, false);
      }
#pragma unroll
      for (int rr = 0; rr < 8; ++rr) {                // C/D layout: M = half*8+rr, N = m
        int M = half * 8 + rr;
        slds[0][M][m] = s0[rr];
        slds[1][M][m] = s1[rr];
      }
      __syncthreads();
      if (lane < 16) {                                // extract banded diagonal
        int i = lane;
#pragma unroll
        for (int dx = 0; dx < WIN; ++dx) {
          int j = i + dx;
          float v = (i < 8) ? slds[0][i][j] : slds[1][i][j - 8];
          simw[i][dy * WIN + dx] = v * INV_T;
        }
      }
      __syncthreads();
    } else {                                          // OOB row: zero-padded logits
      if (lane < 16) {
#pragma unroll
        for (int dx = 0; dx < WIN; ++dx) simw[lane][dy * WIN + dx] = 0.f;
      }
      __syncthreads();
    }
  }

  // ---- softmax over 81 window positions (lane = pixel) ----
  if (lane < 16) {
    float mx = -3.0e38f;
    for (int t = 0; t < WIN * WIN; ++t) mx = fmaxf(mx, simw[lane][t]);
    float s = 0.f;
    for (int t = 0; t < WIN * WIN; ++t) {
      float e = expf(simw[lane][t] - mx);
      simw[lane][t] = e;
      s += e;
    }
    float inv = 1.f / s;
    for (int t = 0; t < WIN * WIN; ++t) simw[lane][t] *= inv;
  }
  __syncthreads();

  // ---- blend phase: aligned[c,i] = sum_j K[c,j] * W[j,i] ----
  v8f acc0 = zero8(), acc1 = zero8();
  for (int dy = 0; dy < WIN; ++dy) {
    int r = h + dy - PADR;
    if (r < 0 || r >= HH) continue;                   // zero key slab -> no contribution
    v16h bw;                                          // banded weight matrix B[K=j][N=i]
#pragma unroll
    for (int v = 0; v < 16; ++v) {
      int j  = v + 16 * half;
      int dx = j - m;
      float wv = (dx >= 0 && dx < WIN) ? simw[m][dy * WIN + dx] : 0.f;
      bw[v] = (_Float16)wv;
    }
    const _Float16* a0 = kn_c + ((size_t)((b * CH + m)      * HH + r)) * WP + w0;
    const _Float16* a1 = kn_c + ((size_t)((b * CH + 16 + m) * HH + r)) * WP + w0;
    v16h ak0 = loadA(a0, half);                       // channels 0..15  x keys j=0..31
    v16h ak1 = loadA(a1, half);                       // channels 16..31 x keys j=0..31
    acc0 = __builtin_amdgcn_wmma_f32_16x16x32_f16(false, ak0, false, bw,
                                                  (short)0, acc0, false, false);
    acc1 = __builtin_amdgcn_wmma_f32_16x16x32_f16(false, ak1, false, bw,
                                                  (short)0, acc1, false, false);
  }

  // write out: D layout lane holds pixel N=m, channel M = half*8 + rr
#pragma unroll
  for (int rr = 0; rr < 8; ++rr) {
    int M = half * 8 + rr;
    out[((size_t)(b * CH + M)      * HH + h) * WW + w0 + m] = acc0[rr];
    out[((size_t)(b * CH + 16 + M) * HH + h) * WW + w0 + m] = acc1[rr];
  }
}

extern "C" void kernel_launch(void* const* d_in, const int* in_sizes, int n_in,
                              void* d_out, int out_size, void* d_ws, size_t ws_size,
                              hipStream_t stream) {
  const float* q = (const float*)d_in[0];
  const float* k = (const float*)d_in[1];
  float* out = (float*)d_out;

  _Float16* qn_t = (_Float16*)d_ws;                 // [B][H][W][C] f16   (4.19 MB)
  _Float16* kn_t = qn_t + QN_T_ELEMS;               // [B][H][W][C] f16   (4.19 MB)
  _Float16* kn_c = kn_t + QN_T_ELEMS;               // [B][C][H][WP] f16  (4.72 MB)

  hipLaunchKernelGGL(SSA_nrm_kernel, dim3(NPIX / 256), dim3(256), 0, stream,
                     q, k, qn_t, kn_t, kn_c);
  hipLaunchKernelGGL(SSA_pad_kernel, dim3(NPAD / 256), dim3(256), 0, stream, kn_c);
  hipLaunchKernelGGL(SSA_align_kernel, dim3(BATCH * HH * (WW / 16)), dim3(32), 0, stream,
                     qn_t, kn_t, kn_c, out);
}
